// OrbitalBackflowNet_18056042512513
// MI455X (gfx1250) — matrix operand order repaired and checked by reading, compiled, not run on hardware
//
#include <hip/hip_runtime.h>
#include <stdint.h>

// Problem constants (match reference)
#define BB   16
#define NN   256
#define HH   64
#define EE   64
#define NOCCC 128

typedef __attribute__((ext_vector_type(16))) _Float16 v16h;
typedef __attribute__((ext_vector_type(8)))  float    v8f;

union AFrag { v16h v; uint32_t u[8]; };
union H2    { _Float16 h[2]; uint32_t u; };

__device__ __forceinline__ uint32_t pack2(float a, float b){
  H2 x; x.h[0] = (_Float16)a; x.h[1] = (_Float16)b; return x.u;
}
__device__ __forceinline__ float silu_f(float v){ return v / (1.0f + expf(-v)); }

// A-fragment (16x32 f16) loaded from a row-major f16 LDS row pointer.
// lane<16 holds K {0..15}+kbase ; lane>=16 holds K {16..31}+kbase
__device__ __forceinline__ v16h loadA_row(const _Float16* row, int g, int kbase){
  AFrag f;
  #pragma unroll
  for (int r = 0; r < 8; ++r){
    const int koff = kbase + g*16 + 2*r;
    f.u[r] = *reinterpret_cast<const uint32_t*>(row + koff);
  }
  return f.v;
}

// B-fragment pre-swizzled in d_ws: [frag][lane][8 x u32]
__device__ __forceinline__ v16h loadB(const uint32_t* ws, int frag, int lane){
  AFrag f;
  const uint4* q = reinterpret_cast<const uint4*>(ws + (size_t)frag*256 + (size_t)lane*8);
  uint4 a = q[0], b = q[1];
  f.u[0]=a.x; f.u[1]=a.y; f.u[2]=a.z; f.u[3]=a.w;
  f.u[4]=b.x; f.u[5]=b.y; f.u[6]=b.z; f.u[7]=b.w;
  return f.v;
}

__device__ __forceinline__ v8f wmma32(v16h a, v16h b, v8f c){
  return __builtin_amdgcn_wmma_f32_16x16x32_f16(false, a, false, b, (short)0, c, false, false);
}

// B-fragment order in workspace:
enum { FB_WE1 = 0, FB_WE2 = 4, FB_WU1 = 12, FB_WU2 = 36, FB_WEV = 44, FB_TOTAL = 52 };

// ---------------------------------------------------------------------------
// Prep: swizzle fp32 weights (stored (K,64) row-major) into f16 WMMA B-frags.
// grid = FB_TOTAL blocks, 256 threads (lane = t>>3, r = t&7).
// ---------------------------------------------------------------------------
__global__ void obf_prep(const float* __restrict__ We1, const float* __restrict__ We2,
                         const float* __restrict__ Wu1, const float* __restrict__ Wu2,
                         const float* __restrict__ W_ev, uint32_t* __restrict__ ws)
{
  const int f = blockIdx.x;
  const int t = threadIdx.x;
  const int lane = t >> 3;
  const int r = t & 7;
  const float* W; int kb, nt, Ksrc;
  if      (f < 4)  { W = We1;  kb = 0;           nt = f;          Ksrc = 5;   }
  else if (f < 12) { W = We2;  kb = (f-4)  >> 2; nt = (f-4)  & 3; Ksrc = 64;  }
  else if (f < 36) { W = Wu1;  kb = (f-12) >> 2; nt = (f-12) & 3; Ksrc = 192; }
  else if (f < 44) { W = Wu2;  kb = (f-36) >> 2; nt = (f-36) & 3; Ksrc = 64;  }
  else             { W = W_ev; kb = (f-44) >> 2; nt = (f-44) & 3; Ksrc = 64;  }
  const int n  = nt*16 + (lane & 15);
  const int k0 = kb*32 + ((lane >> 4) << 4) + 2*r;
  const float a = (k0     < Ksrc) ? W[(size_t)k0     * 64 + n] : 0.0f;
  const float b = (k0 + 1 < Ksrc) ? W[(size_t)(k0+1) * 64 + n] : 0.0f;
  ws[(size_t)f*256 + (size_t)lane*8 + r] = pack2(a, b);
}

// ---------------------------------------------------------------------------
// Main fused kernel: one block = (batch b, i-tile of 16 nodes). 8 waves.
// ---------------------------------------------------------------------------
__launch_bounds__(256, 1)
__global__ void obf_main(
    const float* __restrict__ x,     const int*   __restrict__ spin,
    const float* __restrict__ W_node,const float* __restrict__ b_node,
    const float* __restrict__ be1,   const float* __restrict__ be2,
    const float* __restrict__ W_ve,
    const float* __restrict__ bu1,   const float* __restrict__ bu2,
    const float* __restrict__ Wn1,   const float* __restrict__ bn1,
    const float* __restrict__ Wn2,   const float* __restrict__ bn2,
    const float* __restrict__ W_orb, const float* __restrict__ b_orb,
    const float* __restrict__ bf_raw,
    const uint32_t* __restrict__ wsfrag,
    float* __restrict__ out)
{
  __shared__ __align__(16) _Float16 s_v2e[NN * 66];      // 33792 B, f16 (padded stride)
  __shared__ __align__(16) _Float16 s_scr[8 * 16 * 66];  // 16896 B, per-wave scratch
  __shared__ float s_hvi[16 * 64];                       //  4096 B
  __shared__ float s_mv [16 * 64];                       //  4096 B
  __shared__ float s_bias[4 * 64];                       //  1024 B (be1,be2,bu1,bu2)
  __shared__ float s_x  [NN * 3];                        //  3072 B
  // total 62976 B

  const int t  = threadIdx.x;
  const int bb = blockIdx.x >> 4;
  const int it = blockIdx.x & 15;

  // ---- Phase A0: stage W_ve (as f16, reusing scratch), x slice, biases ----
  _Float16* wve = s_scr; // 64*64 f16 fits in scratch
  for (int idx = t; idx < 64*64; idx += 256) wve[idx] = (_Float16)W_ve[idx];
  for (int idx = t; idx < NN*3;  idx += 256) s_x[idx] = x[(size_t)bb*NN*3 + idx];
  if (t < 64){
    s_bias[t]       = be1[t];
    s_bias[64 + t]  = be2[t];
    s_bias[128 + t] = bu1[t];
    s_bias[192 + t] = bu2[t];
  }
  __syncthreads();

  // ---- Phase A1: h_v and v2e = h_v @ W_ve for all 256 nodes of this batch ----
  {
    const int n = t;
    const float n0 = s_x[n*3+0], n1 = s_x[n*3+1], n2 = s_x[n*3+2];
    const float n3 = (float)spin[n];
    float hv[64];
    #pragma unroll
    for (int j = 0; j < 64; ++j)
      hv[j] = b_node[j] + n0*W_node[0*64+j] + n1*W_node[1*64+j]
                        + n2*W_node[2*64+j] + n3*W_node[3*64+j];
    if ((n >> 4) == it){
      const int il = n & 15;
      #pragma unroll
      for (int j = 0; j < 64; ++j) s_hvi[il*64 + j] = hv[j];
    }
    for (int j = 0; j < 64; ++j){
      float s = 0.0f;
      #pragma unroll
      for (int k = 0; k < 64; ++k) s += hv[k] * (float)wve[k*64 + j];
      s_v2e[n*66 + j] = (_Float16)s;
    }
  }
  __syncthreads();

  // ---- Phase B: fused edge pipeline, WMMA chains. Wave owns 2 i rows. ----
  const int lane = t & 31;
  const int wv   = t >> 5;
  const int g    = lane >> 4;
  const int m    = lane & 15;
  _Float16* scr = s_scr + wv * 16 * 66;        // wave-private 16x64 (stride 66) tile
  const _Float16* scrrow = scr + m * 66;

  for (int ii = 0; ii < 2; ++ii){
    const int il = wv*2 + ii;
    const int ig = it*16 + il;
    const float xi0 = s_x[ig*3+0], xi1 = s_x[ig*3+1], xi2 = s_x[ig*3+2];
    const _Float16* vrow_i = s_v2e + ig*66;
    float macc[4] = {0.f, 0.f, 0.f, 0.f};

    for (int jt = 0; jt < 16; ++jt){
      const int j = jt*16 + m;
      const _Float16* vrow_j = s_v2e + j*66;

      // edge_in A fragment: K = [rx,ry,rz,|r|,r^2, 0...] padded to 32
      AFrag ae;
      #pragma unroll
      for (int r = 0; r < 8; ++r) ae.u[r] = 0u;
      if (g == 0){
        const float rx = xi0 - s_x[j*3+0];
        const float ry = xi1 - s_x[j*3+1];
        const float rz = xi2 - s_x[j*3+2];
        const float r2 = rx*rx + ry*ry + rz*rz;
        const float r1 = sqrtf(r2 + 1e-12f);
        ae.u[0] = pack2(rx, ry);
        ae.u[1] = pack2(rz, r1);
        ae.u[2] = pack2(r2, 0.0f);
      }

      v8f c[4];

      // GEMM1: silu(edge_in @ We1 + be1)
      #pragma unroll
      for (int nt = 0; nt < 4; ++nt){
        const float bv = s_bias[0*64 + nt*16 + m];
        #pragma unroll
        for (int v = 0; v < 8; ++v) c[nt][v] = bv;
        c[nt] = wmma32(ae.v, loadB(wsfrag, FB_WE1 + nt, lane), c[nt]);
      }
      #pragma unroll
      for (int nt = 0; nt < 4; ++nt)
        #pragma unroll
        for (int v = 0; v < 8; ++v)
          scr[(v + 8*g)*66 + nt*16 + m] = (_Float16)silu_f(c[nt][v]);

      // GEMM2: h_e = (.) @ We2 + be2
      #pragma unroll
      for (int nt = 0; nt < 4; ++nt){
        const float bv = s_bias[1*64 + nt*16 + m];
        #pragma unroll
        for (int v = 0; v < 8; ++v) c[nt][v] = bv;
      }
      #pragma unroll
      for (int kb = 0; kb < 2; ++kb){
        const v16h a = loadA_row(scrrow, g, kb*32);
        #pragma unroll
        for (int nt = 0; nt < 4; ++nt)
          c[nt] = wmma32(a, loadB(wsfrag, FB_WE2 + kb*4 + nt, lane), c[nt]);
      }
      #pragma unroll
      for (int nt = 0; nt < 4; ++nt)
        #pragma unroll
        for (int v = 0; v < 8; ++v)
          scr[(v + 8*g)*66 + nt*16 + m] = (_Float16)c[nt][v];

      // GEMM3: silu([h_e, v_i, v_j] @ Wu1 + bu1), K = 192
      #pragma unroll
      for (int nt = 0; nt < 4; ++nt){
        const float bv = s_bias[2*64 + nt*16 + m];
        #pragma unroll
        for (int v = 0; v < 8; ++v) c[nt][v] = bv;
      }
      #pragma unroll
      for (int kb = 0; kb < 6; ++kb){
        const _Float16* arow = (kb < 2) ? scrrow : ((kb < 4) ? vrow_i : vrow_j);
        const int kbase = ((kb < 2) ? kb : ((kb < 4) ? (kb-2) : (kb-4))) * 32;
        const v16h a = loadA_row(arow, g, kbase);
        #pragma unroll
        for (int nt = 0; nt < 4; ++nt)
          c[nt] = wmma32(a, loadB(wsfrag, FB_WU1 + kb*4 + nt, lane), c[nt]);
      }
      #pragma unroll
      for (int nt = 0; nt < 4; ++nt)
        #pragma unroll
        for (int v = 0; v < 8; ++v)
          scr[(v + 8*g)*66 + nt*16 + m] = (_Float16)silu_f(c[nt][v]);

      // GEMM4: h_e_new = (.) @ Wu2 + bu2
      #pragma unroll
      for (int nt = 0; nt < 4; ++nt){
        const float bv = s_bias[3*64 + nt*16 + m];
        #pragma unroll
        for (int v = 0; v < 8; ++v) c[nt][v] = bv;
      }
      #pragma unroll
      for (int kb = 0; kb < 2; ++kb){
        const v16h a = loadA_row(scrrow, g, kb*32);
        #pragma unroll
        for (int nt = 0; nt < 4; ++nt)
          c[nt] = wmma32(a, loadB(wsfrag, FB_WU2 + kb*4 + nt, lane), c[nt]);
      }
      #pragma unroll
      for (int nt = 0; nt < 4; ++nt)
        #pragma unroll
        for (int v = 0; v < 8; ++v)
          scr[(v + 8*g)*66 + nt*16 + m] = (_Float16)c[nt][v];

      // GEMM5: msgs = h_e_new @ W_ev
      #pragma unroll
      for (int nt = 0; nt < 4; ++nt)
        #pragma unroll
        for (int v = 0; v < 8; ++v) c[nt][v] = 0.0f;
      #pragma unroll
      for (int kb = 0; kb < 2; ++kb){
        const v16h a = loadA_row(scrrow, g, kb*32);
        #pragma unroll
        for (int nt = 0; nt < 4; ++nt)
          c[nt] = wmma32(a, loadB(wsfrag, FB_WEV + kb*4 + nt, lane), c[nt]);
      }

      // diagonal mask (no self edges) + row-sum accumulation
      if (jt == (ig >> 4)){
        const int vd = (ig & 15) - 8*g;
        if (vd >= 0 && vd < 8){
          #pragma unroll
          for (int nt = 0; nt < 4; ++nt) c[nt][vd] = 0.0f;
        }
      }
      #pragma unroll
      for (int nt = 0; nt < 4; ++nt){
        float s = 0.0f;
        #pragma unroll
        for (int v = 0; v < 8; ++v) s += c[nt][v];
        macc[nt] += s;
      }
    } // jt

    #pragma unroll
    for (int nt = 0; nt < 4; ++nt){
      const float tot = macc[nt] + __shfl_xor(macc[nt], 16);
      if (lane < 16) s_mv[il*64 + nt*16 + m] = tot;
    }
  } // ii
  __syncthreads();

  // ---- Phase C: node update + orbital head (fp32 VALU, tiny) ----
  const float spl = log1pf(expf(bf_raw[0]));   // softplus(bf_scale_raw)
  {
    const int il = t >> 4;
    const int cc = t & 15;
    float* tmp1 = reinterpret_cast<float*>(s_scr);   // reuse scratch as fp32 16x64

    for (int jj = 0; jj < 4; ++jj){
      const int jc = cc*4 + jj;
      float s = bn1[jc];
      for (int k = 0; k < 64; ++k) s += s_hvi[il*64 + k] * Wn1[(size_t)k*64 + jc];
      for (int k = 0; k < 64; ++k) s += s_mv [il*64 + k] * Wn1[(size_t)(64 + k)*64 + jc];
      tmp1[il*64 + jc] = silu_f(s);
    }
    __syncthreads();

    for (int jj = 0; jj < 4; ++jj){
      const int jc = cc*4 + jj;
      float s = bn2[jc];
      for (int k = 0; k < 64; ++k) s += tmp1[il*64 + k] * Wn2[(size_t)k*64 + jc];
      s_mv[il*64 + jc] = s_hvi[il*64 + jc] + s;   // h = h_v + delta_h (overwrite m_v)
    }
    __syncthreads();

    const int ig = it*16 + il;
    for (int oo = 0; oo < 8; ++oo){
      const int o = cc*8 + oo;
      float s = b_orb[o];
      for (int k = 0; k < 64; ++k) s += s_mv[il*64 + k] * W_orb[(size_t)k*NOCCC + o];
      out[((size_t)(bb*NN + ig))*NOCCC + o] = tanhf(s) * spl;
    }
  }
}

// ---------------------------------------------------------------------------
extern "C" void kernel_launch(void* const* d_in, const int* in_sizes, int n_in,
                              void* d_out, int out_size, void* d_ws, size_t ws_size,
                              hipStream_t stream)
{
  (void)in_sizes; (void)n_in; (void)out_size; (void)ws_size;
  const float* x      = (const float*)d_in[0];
  const int*   spin   = (const int*)  d_in[1];
  const float* W_node = (const float*)d_in[2];
  const float* b_node = (const float*)d_in[3];
  const float* We1    = (const float*)d_in[4];
  const float* be1    = (const float*)d_in[5];
  const float* We2    = (const float*)d_in[6];
  const float* be2    = (const float*)d_in[7];
  const float* W_ve   = (const float*)d_in[8];
  const float* W_ev   = (const float*)d_in[9];
  const float* Wu1    = (const float*)d_in[10];
  const float* bu1    = (const float*)d_in[11];
  const float* Wu2    = (const float*)d_in[12];
  const float* bu2    = (const float*)d_in[13];
  const float* Wn1    = (const float*)d_in[14];
  const float* bn1    = (const float*)d_in[15];
  const float* Wn2    = (const float*)d_in[16];
  const float* bn2    = (const float*)d_in[17];
  const float* W_orb  = (const float*)d_in[18];
  const float* b_orb  = (const float*)d_in[19];
  const float* bf_raw = (const float*)d_in[20];

  uint32_t* ws  = (uint32_t*)d_ws;   // needs FB_TOTAL*256*4 = 53248 bytes
  float*    out = (float*)d_out;

  obf_prep<<<FB_TOTAL, 256, 0, stream>>>(We1, We2, Wu1, Wu2, W_ev, ws);
  obf_main<<<BB * (NN/16), 256, 0, stream>>>(x, spin, W_node, b_node, be1, be2, W_ve,
                                             bu1, bu2, Wn1, bn1, Wn2, bn2,
                                             W_orb, b_orb, bf_raw, ws, out);
}
